// SPFNSAttentionHead_89575837925541
// MI455X (gfx1250) — matrix operand order repaired and checked by reading, not compiled, not run on hardware
//
#include <hip/hip_runtime.h>
#include <hip/hip_bf16.h>

// ---- problem constants (match reference) ----
#define SEQ   4096
#define BATCH 4
#define HID   512
#define HD    64
#define EPS_CLIP 1e-7f
#define NEG_EXP  (-3.5f)   // -(D_INTRINSIC + ALPHA)

typedef __attribute__((ext_vector_type(16))) _Float16 v16h;
typedef __attribute__((ext_vector_type(8)))  _Float16 v8h;
typedef __attribute__((ext_vector_type(8)))  float    v8f;

// ---------------------------------------------------------------------------
// Fragment loaders (CDNA5 wave32 WMMA 16x16x32 f16; contiguous k order used here for A and B, not AMD's table order)
// A 16x32: lane L = row L%16; e -> K=k0+16*half+e  (row contiguous in memory)
// B 32x16: lane L = col L%16; e -> K=k0+16*half+e  (column contiguous in memory)
// ---------------------------------------------------------------------------
__device__ inline v16h load_fragA_f16(const _Float16* __restrict__ rowptr, int k0, int half) {
  return *(const v16h*)(rowptr + k0 + 16 * half);
}
__device__ inline v16h load_fragB_f16(const _Float16* __restrict__ colptr, int k0, int half) {
  return *(const v16h*)(colptr + k0 + 16 * half);  // 32B contiguous
}

// ---------------------------------------------------------------------------
// Kernel 1: q/k/v projection + bias + L2 normalize. One wave per token row.
// Emits f16 q,k (row-major [B*S,64]) and f16 v TRANSPOSED ([B,64,S]) so PV
// B-fragments are contiguous.
// ---------------------------------------------------------------------------
__global__ __launch_bounds__(256) void spfns_proj_kernel(
    const float* __restrict__ x,
    const float* __restrict__ Wq, const float* __restrict__ bq,
    const float* __restrict__ Wk, const float* __restrict__ bk,
    const float* __restrict__ Wv, const float* __restrict__ bv,
    _Float16* __restrict__ qh, _Float16* __restrict__ kh, _Float16* __restrict__ vth) {
  __shared__ float xs[8][HID];
  const int wave = threadIdx.x >> 5;
  const int lane = threadIdx.x & 31;
  const int row  = blockIdx.x * 8 + wave;               // [0, B*S)
  const float* xr = x + (size_t)row * HID;
  for (int h = lane; h < HID; h += 32) xs[wave][h] = xr[h];
  __syncthreads();

  const int d0 = lane, d1 = lane + 32;
  float aq0 = bq[d0], aq1 = bq[d1];
  float ak0 = bk[d0], ak1 = bk[d1];
  float av0 = bv[d0], av1 = bv[d1];
  for (int h = 0; h < HID; ++h) {
    const float xv = xs[wave][h];
    aq0 = fmaf(xv, Wq[h * HD + d0], aq0);
    aq1 = fmaf(xv, Wq[h * HD + d1], aq1);
    ak0 = fmaf(xv, Wk[h * HD + d0], ak0);
    ak1 = fmaf(xv, Wk[h * HD + d1], ak1);
    av0 = fmaf(xv, Wv[h * HD + d0], av0);
    av1 = fmaf(xv, Wv[h * HD + d1], av1);
  }
  float nq = aq0 * aq0 + aq1 * aq1;
  float nk = ak0 * ak0 + ak1 * ak1;
  float nv = av0 * av0 + av1 * av1;
#pragma unroll
  for (int m = 16; m >= 1; m >>= 1) {
    nq += __shfl_xor(nq, m, 32);
    nk += __shfl_xor(nk, m, 32);
    nv += __shfl_xor(nv, m, 32);
  }
  const float iq = 1.0f / fmaxf(sqrtf(nq), 1e-12f);
  const float ik = 1.0f / fmaxf(sqrtf(nk), 1e-12f);
  const float iv = 1.0f / fmaxf(sqrtf(nv), 1e-12f);

  _Float16* qrow = qh + (size_t)row * HD;
  _Float16* krow = kh + (size_t)row * HD;
  qrow[d0] = (_Float16)(aq0 * iq);  qrow[d1] = (_Float16)(aq1 * iq);
  krow[d0] = (_Float16)(ak0 * ik);  krow[d1] = (_Float16)(ak1 * ik);
  const int b = row / SEQ, s = row % SEQ;
  vth[((size_t)(b * HD + d0)) * SEQ + s] = (_Float16)(av0 * iv);
  vth[((size_t)(b * HD + d1)) * SEQ + s] = (_Float16)(av1 * iv);
}

// ---------------------------------------------------------------------------
// Kernel 2: one wave per 16x16 score tile. cos = q.k^T via two WMMA (K=64),
// then clip / arccos / power-law, raw score stored into the probs region.
// ---------------------------------------------------------------------------
__global__ __launch_bounds__(32) void spfns_score_kernel(
    const _Float16* __restrict__ qh, const _Float16* __restrict__ kh,
    float* __restrict__ probs) {
  const int t0 = blockIdx.x * 16;
  const int s0 = blockIdx.y * 16;
  const int b  = blockIdx.z;
  const int lane = threadIdx.x;
  const int idx  = lane & 15;
  const int half = lane >> 4;

  const _Float16* qrow = qh + (size_t)(b * SEQ + s0 + idx) * HD;  // A row
  const _Float16* kcol = kh + (size_t)(b * SEQ + t0 + idx) * HD;  // B column (= k row)

  v16h a0 = load_fragA_f16(qrow, 0, half);
  v16h a1 = load_fragA_f16(qrow, 32, half);
  v16h b0 = load_fragB_f16(kcol, 0, half);
  v16h b1 = load_fragB_f16(kcol, 32, half);

  v8f c = {};
  c = __builtin_amdgcn_wmma_f32_16x16x32_f16(false, a0, false, b0, (short)0, c, false, false);
  c = __builtin_amdgcn_wmma_f32_16x16x32_f16(false, a1, false, b1, (short)0, c, false, false);

#pragma unroll
  for (int r = 0; r < 8; ++r) {
    float cosv = fminf(fmaxf(c[r], -1.0f + EPS_CLIP), 1.0f - EPS_CLIP);
    float g    = acosf(cosv);                        // geodesic distance (R=1)
    float sc   = __expf(NEG_EXP * __logf(1.0f + g)); // (1+g)^-3.5
    const int srow = s0 + r + 8 * half;
    probs[(size_t)(b * SEQ + srow) * SEQ + (t0 + idx)] = sc;
  }
}

// ---------------------------------------------------------------------------
// Kernel 3: deterministic column sums N_C[t] = sum_s score[s][t].
// Coalesced: consecutive threads read consecutive t.
// ---------------------------------------------------------------------------
__global__ __launch_bounds__(256) void spfns_colsum_kernel(
    const float* __restrict__ probs, float* __restrict__ NC) {
  const int b = blockIdx.y;
  const int t = blockIdx.x * 256 + threadIdx.x;
  const float* base = probs + (size_t)b * SEQ * SEQ + t;
  float acc = 0.0f;
  for (int s = 0; s < SEQ; ++s) acc += base[(size_t)s * SEQ];
  NC[b * SEQ + t] = acc;
}

// ---------------------------------------------------------------------------
// Kernel 4: per 16-row tile: M(s) = sum_t score*rsqrt(NC); then in-place
// probs = score*rsqrt(NC)/M(s) and attention_output = probs @ v via WMMA.
// 4 waves/block; waves split the t dimension, 4 d-tile accumulators each,
// cross-wave reduction through LDS.
// ---------------------------------------------------------------------------
__global__ __launch_bounds__(128) void spfns_rescale_pv_kernel(
    const _Float16* __restrict__ vth, const float* __restrict__ NC,
    float* __restrict__ probs, float* __restrict__ outAttn) {
  __shared__ float Ms[16];
  __shared__ float red[4][4][8][32];  // [wave][dtile][reg][lane]

  const int b    = blockIdx.y;
  const int s0   = blockIdx.x * 16;
  const int wave = threadIdx.x >> 5;
  const int lane = threadIdx.x & 31;
  const int idx  = lane & 15;
  const int half = lane >> 4;
  const float* ncb = NC + b * SEQ;

  // Phase A: row scale M(s) for the 16 rows (wave w handles rows w, w+4, ...)
  for (int rr = wave; rr < 16; rr += 4) {
    const float* prow = probs + (size_t)(b * SEQ + s0 + rr) * SEQ;
    float acc = 0.0f;
    for (int t = lane; t < SEQ; t += 32)
      acc += prow[t] * rsqrtf(fmaxf(ncb[t], 1e-30f));
#pragma unroll
    for (int m = 16; m >= 1; m >>= 1) acc += __shfl_xor(acc, m, 32);
    if (lane == 0) Ms[rr] = acc;
  }
  __syncthreads();

  // Phase B: rescale + PV
  const float invM = 1.0f / fmaxf(Ms[idx], 1e-12f);   // A-fragment row of this lane
  float* prow = probs + (size_t)(b * SEQ + s0 + idx) * SEQ;
  v8f acc[4] = {};

  for (int ch = wave; ch < SEQ / 32; ch += 4) {
    const int t0 = ch * 32;
    // raw scores for this lane's A-fragment elements (two 32B runs)
    v8f sA = *(const v8f*)(prow + t0 + 8 * half);
    v8f sB = *(const v8f*)(prow + t0 + 16 + 8 * half);
    v8f pA, pB;
    v16h a;
#pragma unroll
    for (int i = 0; i < 8; ++i) {
      const int ta = t0 + 8 * half + i;
      const int tb = t0 + 16 + 8 * half + i;
      const float va = sA[i] * rsqrtf(fmaxf(ncb[ta], 1e-30f)) * invM;
      const float vb = sB[i] * rsqrtf(fmaxf(ncb[tb], 1e-30f)) * invM;
      pA[i] = va; pB[i] = vb;
      a[i] = (_Float16)va; a[8 + i] = (_Float16)vb;
    }
    // write final attention_probs in place
    *(v8f*)(prow + t0 + 8 * half) = pA;
    *(v8f*)(prow + t0 + 16 + 8 * half) = pB;
    // accumulate P @ V over 4 d-tiles (v stored transposed: columns contiguous)
#pragma unroll
    for (int j = 0; j < 4; ++j) {
      const _Float16* vcol = vth + (size_t)(b * HD + 16 * j + idx) * SEQ;
      v8h blo = *(const v8h*)(vcol + t0 + 8 * half);
      v8h bhi = *(const v8h*)(vcol + t0 + 16 + 8 * half);
      v16h bf;
#pragma unroll
      for (int i = 0; i < 8; ++i) { bf[i] = blo[i]; bf[8 + i] = bhi[i]; }
      acc[j] = __builtin_amdgcn_wmma_f32_16x16x32_f16(false, a, false, bf, (short)0,
                                                      acc[j], false, false);
    }
  }

  // cross-wave reduction
#pragma unroll
  for (int j = 0; j < 4; ++j)
#pragma unroll
    for (int r = 0; r < 8; ++r) red[wave][j][r][lane] = acc[j][r];
  __syncthreads();

  // wave w finalizes d-tile j = w
#pragma unroll
  for (int r = 0; r < 8; ++r) {
    float sum = red[0][wave][r][lane] + red[1][wave][r][lane] +
                red[2][wave][r][lane] + red[3][wave][r][lane];
    const int s = s0 + r + 8 * half;
    const int d = 16 * wave + idx;
    outAttn[(size_t)(b * SEQ + s) * HD + d] = sum;
  }
}

// ---------------------------------------------------------------------------
extern "C" void kernel_launch(void* const* d_in, const int* in_sizes, int n_in,
                              void* d_out, int out_size, void* d_ws, size_t ws_size,
                              hipStream_t stream) {
  const float* x  = (const float*)d_in[0];
  const float* Wq = (const float*)d_in[1];
  const float* bq = (const float*)d_in[2];
  const float* Wk = (const float*)d_in[3];
  const float* bk = (const float*)d_in[4];
  const float* Wv = (const float*)d_in[5];
  const float* bv = (const float*)d_in[6];

  float* outAttn = (float*)d_out;                              // [B,1,S,64]
  float* probs   = outAttn + (size_t)BATCH * SEQ * HD;         // [B,1,S,S]

  const size_t QKV_BYTES = (size_t)BATCH * SEQ * HD * sizeof(_Float16);  // 2 MB each
  char* ws = (char*)d_ws;
  _Float16* qh  = (_Float16*)(ws);
  _Float16* kh  = (_Float16*)(ws + QKV_BYTES);
  _Float16* vth = (_Float16*)(ws + 2 * QKV_BYTES);
  float*    NC  = (float*)(ws + 3 * QKV_BYTES);                // [B,S]

  spfns_proj_kernel<<<(BATCH * SEQ) / 8, 256, 0, stream>>>(
      x, Wq, bq, Wk, bk, Wv, bv, qh, kh, vth);

  spfns_score_kernel<<<dim3(SEQ / 16, SEQ / 16, BATCH), 32, 0, stream>>>(
      qh, kh, probs);

  spfns_colsum_kernel<<<dim3(SEQ / 256, BATCH), 256, 0, stream>>>(probs, NC);

  spfns_rescale_pv_kernel<<<dim3(SEQ / 16, BATCH), 128, 0, stream>>>(
      vth, NC, probs, outAttn);
}
